// DisenGCNLayer_75857712382346
// MI455X (gfx1250) — matrix operand order repaired and checked by reading, not compiled, not run on hardware
//
#include <hip/hip_runtime.h>
#include <hip/hip_fp16.h>

typedef __attribute__((ext_vector_type(16))) _Float16 v16h;
typedef __attribute__((ext_vector_type(8)))  float    v8f;

#define N_NODES 50000
#define N_EDGES 800000
#define FEATS   128
#define KF      8      // factors
#define FD      16     // factor dim
#define ITERS   4

// ---------------------------------------------------------------------------
// Float atomic max via ordered-int trick (compiles everywhere, works with
// -inf init: positives compare as signed ints, negatives reversed as unsigned)
// ---------------------------------------------------------------------------
__device__ __forceinline__ void atomic_max_float(float* addr, float v) {
  int iv = __float_as_int(v);
  if (iv >= 0) {
    atomicMax((int*)addr, iv);
  } else {
    atomicMin((unsigned int*)addr, (unsigned int)iv);
  }
}

// ---------------------------------------------------------------------------
// W[128][128] f32 row-major -> Wt[n][k] f16 (column-contiguous for B frags)
// ---------------------------------------------------------------------------
__global__ void wt_kernel(const float* __restrict__ W, _Float16* __restrict__ Wt) {
  int idx = blockIdx.x * 256 + threadIdx.x;      // idx = n*128 + k
  if (idx >= FEATS * FEATS) return;
  int n = idx >> 7, k = idx & 127;
  Wt[idx] = (_Float16)W[k * FEATS + n];
}

// ---------------------------------------------------------------------------
// h = leaky_relu(x @ W + b); per-factor L2 normalize; write h_normed and
// initial h_dst. One block = 16 rows; wave w = factor w (N-tile w*16).
// A frag: lane(m,hi) holds 16 contiguous K = kb+hi*16+0..15.
// B frag: lane(n=m,hi) holds 16 contiguous K from Wt[col] at kb+hi*16.
// ---------------------------------------------------------------------------
__global__ void __launch_bounds__(256)
gemm_norm_kernel(const float* __restrict__ x, const _Float16* __restrict__ Wt,
                 const float* __restrict__ bias,
                 float* __restrict__ hsrc, float* __restrict__ hdst) {
  const int lane = threadIdx.x & 31;
  const int wv   = threadIdx.x >> 5;   // 0..7 -> factor / N-tile
  const int m    = lane & 15;
  const int hi   = lane >> 4;
  const int row0 = blockIdx.x * 16;
  const int col  = wv * 16 + m;

  const float*    ap = x  + (size_t)(row0 + m) * FEATS;
  const _Float16* bp = Wt + (size_t)col * FEATS + hi * 16;

  v8f c = {};
#pragma unroll
  for (int kb = 0; kb < FEATS; kb += 32) {
    float4 a0 = *(const float4*)(ap + kb + hi * 16);
    float4 a1 = *(const float4*)(ap + kb + hi * 16 + 4);
    float4 a2 = *(const float4*)(ap + kb + hi * 16 + 8);
    float4 a3 = *(const float4*)(ap + kb + hi * 16 + 12);
    v16h a;
    a[0] = (_Float16)a0.x; a[1] = (_Float16)a0.y; a[2]  = (_Float16)a0.z; a[3]  = (_Float16)a0.w;
    a[4] = (_Float16)a1.x; a[5] = (_Float16)a1.y; a[6]  = (_Float16)a1.z; a[7]  = (_Float16)a1.w;
    a[8] = (_Float16)a2.x; a[9] = (_Float16)a2.y; a[10] = (_Float16)a2.z; a[11] = (_Float16)a2.w;
    a[12]= (_Float16)a3.x; a[13]= (_Float16)a3.y; a[14] = (_Float16)a3.z; a[15] = (_Float16)a3.w;
    v16h b = *(const v16h*)(bp + kb);
    c = __builtin_amdgcn_wmma_f32_16x16x32_f16(false, a, false, b,
                                               (short)0, c, false, false);
  }

  const float bv = bias[col];
#pragma unroll
  for (int j = 0; j < 8; ++j) {
    float v = c[j] + bv;
    v = v > 0.0f ? v : 0.01f * v;               // leaky_relu
    float s = v * v;                            // sum over the 16 cols (lanes)
    s += __shfl_xor(s, 1, 32);
    s += __shfl_xor(s, 2, 32);
    s += __shfl_xor(s, 4, 32);
    s += __shfl_xor(s, 8, 32);
    v *= rsqrtf(s);
    size_t o = (size_t)(row0 + j + 8 * hi) * FEATS + col;
    hsrc[o] = v;
    hdst[o] = v;
  }
}

// ---------------------------------------------------------------------------
// Per-iteration state init: m=-inf, denom=0, agg=0
// ---------------------------------------------------------------------------
__global__ void init_kernel(float* __restrict__ mx, float* __restrict__ denom,
                            float* __restrict__ agg) {
  int idx = blockIdx.x * 256 + threadIdx.x;
  const int NK = N_NODES * KF;
  if (idx < NK) {
    mx[idx] = -__builtin_huge_valf();
  } else if (idx < 2 * NK) {
    denom[idx - NK] = 0.0f;
  } else {
    int j = idx - 2 * NK;
    if (j < N_NODES * FEATS) agg[j] = 0.0f;
  }
}

// ---------------------------------------------------------------------------
// Wave per edge: scores[e][k] = <h_dst[row], h_src[col]>_factor-k, atomic max.
// Lane l owns feats 4l..4l+3 (all inside factor l>>2); quad butterfly reduces.
// ---------------------------------------------------------------------------
__global__ void __launch_bounds__(256)
score_kernel(const long long* __restrict__ ei, const float* __restrict__ hdst,
             const float* __restrict__ hsrc, float* __restrict__ scores,
             float* __restrict__ mx) {
  int wid = (blockIdx.x * 256 + threadIdx.x) >> 5;
  if (wid >= N_EDGES) return;
  int lane = threadIdx.x & 31;
  long long r = ei[wid];
  long long cc = ei[N_EDGES + wid];
  float4 hd = *(const float4*)(hdst + (size_t)r  * FEATS + 4 * lane);
  float4 hs = *(const float4*)(hsrc + (size_t)cc * FEATS + 4 * lane);
  float p = hd.x * hs.x + hd.y * hs.y + hd.z * hs.z + hd.w * hs.w;
  p += __shfl_xor(p, 1, 32);
  p += __shfl_xor(p, 2, 32);
  if ((lane & 3) == 0) {
    int k = lane >> 2;
    scores[(size_t)wid * KF + k] = p;           // TAU == 1.0
    atomic_max_float(&mx[(size_t)r * KF + k], p);
  }
}

// ---------------------------------------------------------------------------
// ex = exp(score - m[row]); denom[row] += ex (overwrite scores with ex)
// ---------------------------------------------------------------------------
__global__ void exp_kernel(const long long* __restrict__ ei,
                           float* __restrict__ scores,
                           const float* __restrict__ mx,
                           float* __restrict__ denom) {
  int idx = blockIdx.x * 256 + threadIdx.x;
  if (idx >= N_EDGES * KF) return;
  int e = idx >> 3, k = idx & 7;
  long long r = ei[e];
  float ex = __expf(scores[idx] - mx[(size_t)r * KF + k]);
  scores[idx] = ex;
  atomicAdd(&denom[(size_t)r * KF + k], ex);
}

// ---------------------------------------------------------------------------
// Wave per edge: agg[row] += h_src[col] * alpha  (alpha = ex/denom)
// ---------------------------------------------------------------------------
__global__ void __launch_bounds__(256)
agg_kernel(const long long* __restrict__ ei, const float* __restrict__ scores,
           const float* __restrict__ denom, const float* __restrict__ hsrc,
           float* __restrict__ agg) {
  int wid = (blockIdx.x * 256 + threadIdx.x) >> 5;
  if (wid >= N_EDGES) return;
  int lane = threadIdx.x & 31;
  long long r = ei[wid];
  long long cc = ei[N_EDGES + wid];
  int k = lane >> 2;
  float alpha = scores[(size_t)wid * KF + k] / denom[(size_t)r * KF + k];
  float4 hs = *(const float4*)(hsrc + (size_t)cc * FEATS + 4 * lane);
  float* ap = agg + (size_t)r * FEATS + 4 * lane;
  atomicAdd(ap + 0, hs.x * alpha);
  atomicAdd(ap + 1, hs.y * alpha);
  atomicAdd(ap + 2, hs.z * alpha);
  atomicAdd(ap + 3, hs.w * alpha);
}

// ---------------------------------------------------------------------------
// node_attr = agg + h_normed; per-factor renormalize -> h_dst.
// Feature groups of 16 align with 16-lane shuffle halves.
// ---------------------------------------------------------------------------
__global__ void node_kernel(const float* __restrict__ agg,
                            const float* __restrict__ hsrc,
                            float* __restrict__ hdst) {
  int idx = blockIdx.x * 256 + threadIdx.x;
  if (idx >= N_NODES * FEATS) return;
  float v = agg[idx] + hsrc[idx];
  float s = v * v;
  s += __shfl_xor(s, 1, 32);
  s += __shfl_xor(s, 2, 32);
  s += __shfl_xor(s, 4, 32);
  s += __shfl_xor(s, 8, 32);
  v *= rsqrtf(s);
  hdst[idx] = v;
}

// ---------------------------------------------------------------------------
extern "C" void kernel_launch(void* const* d_in, const int* in_sizes, int n_in,
                              void* d_out, int out_size, void* d_ws, size_t ws_size,
                              hipStream_t stream) {
  (void)in_sizes; (void)n_in; (void)out_size; (void)ws_size;
  const float*     x    = (const float*)d_in[0];
  const long long* ei   = (const long long*)d_in[1];   // int64 [2, E]
  const float*     W    = (const float*)d_in[2];
  const float*     bias = (const float*)d_in[3];
  float* out = (float*)d_out;                           // doubles as h_dst

  char* ws = (char*)d_ws;
  size_t off = 0;
  auto alloc = [&](size_t bytes) -> void* {
    void* p = (void*)(ws + off);
    off += (bytes + 255) & ~(size_t)255;
    return p;
  };
  _Float16* Wt     = (_Float16*)alloc((size_t)FEATS * FEATS * sizeof(_Float16));
  float*    hsrc   = (float*)alloc((size_t)N_NODES * FEATS * sizeof(float));
  float*    scores = (float*)alloc((size_t)N_EDGES * KF * sizeof(float));
  float*    mx     = (float*)alloc((size_t)N_NODES * KF * sizeof(float));
  float*    denom  = (float*)alloc((size_t)N_NODES * KF * sizeof(float));
  float*    agg    = (float*)alloc((size_t)N_NODES * FEATS * sizeof(float));

  // h = leaky_relu(x@W+b), factor-normalized -> hsrc and initial h_dst(out)
  wt_kernel<<<(FEATS * FEATS + 255) / 256, 256, 0, stream>>>(W, Wt);
  gemm_norm_kernel<<<N_NODES / 16, 256, 0, stream>>>(x, Wt, bias, hsrc, out);

  const int init_elems = 2 * N_NODES * KF + N_NODES * FEATS;
  for (int it = 0; it < ITERS; ++it) {
    init_kernel<<<(init_elems + 255) / 256, 256, 0, stream>>>(mx, denom, agg);
    score_kernel<<<N_EDGES / 8, 256, 0, stream>>>(ei, out, hsrc, scores, mx);
    exp_kernel<<<(N_EDGES * KF + 255) / 256, 256, 0, stream>>>(ei, scores, mx, denom);
    agg_kernel<<<N_EDGES / 8, 256, 0, stream>>>(ei, scores, denom, hsrc, agg);
    node_kernel<<<(N_NODES * FEATS + 255) / 256, 256, 0, stream>>>(agg, hsrc, out);
  }
}
